// patchnumcl_77653008711688
// MI455X (gfx1250) — matrix operand order repaired and checked by reading, not compiled, not run on hardware
//
#include <hip/hip_runtime.h>
#include <hip/hip_bf16.h>
#include <math.h>

typedef __attribute__((ext_vector_type(16))) __bf16 v16bf;
typedef __attribute__((ext_vector_type(8)))  __bf16 v8bf;
typedef __attribute__((ext_vector_type(8)))  float  v8f;
typedef __attribute__((ext_vector_type(4)))  float  v4f;

#define BB  32
#define NN  128
#define PP  8
#define DD  512
#define MTOT (BB*NN)          // 4096 rows per patch
#define BLK_M 128
#define BLK_N 128

// ---------------- weight fp32 -> bf16 conversion ----------------
__global__ void cvt_weights(const float* __restrict__ w1,
                            const float* __restrict__ w2,
                            __bf16* __restrict__ w1b,
                            __bf16* __restrict__ w2b, int n) {
    int i = blockIdx.x * blockDim.x + threadIdx.x;
    if (i < n) {
        w1b[i] = (__bf16)w1[i];
        w2b[i] = (__bf16)w2[i];
    }
}

// ---------------- x fp32 -> bf16 conversion (8 elems / thread) ----------------
__global__ void cvt_x(const float* __restrict__ x, __bf16* __restrict__ xb,
                      unsigned n8) {
    unsigned i = blockIdx.x * blockDim.x + threadIdx.x;
    if (i < n8) {
        const size_t o = (size_t)i * 8;
        v4f f0 = *(const v4f*)(x + o);
        v4f f1 = *(const v4f*)(x + o + 4);
        v8bf v;
        #pragma unroll
        for (int j = 0; j < 4; ++j) { v[j] = (__bf16)f0[j]; v[4 + j] = (__bf16)f1[j]; }
        *(v8bf*)(xb + o) = v;
    }
}

// B fragment: lane (l,hl) = one column, K {0..7,16..23} (+8 for half 1)
__device__ __forceinline__ v16bf loadB(const __bf16* wp) {
    v8bf lo = *(const v8bf*)(wp);
    v8bf hi = *(const v8bf*)(wp + 16);
    v16bf b;
    #pragma unroll
    for (int i = 0; i < 8; ++i) { b[i] = lo[i]; b[8 + i] = hi[i]; }
    return b;
}

// A fragment from fp32 row: half 0 -> K {0..7,16..23}, half 1 -> +8
__device__ __forceinline__ v16bf loadA_f32(const float* row, int hl) {
    v4f a0 = *(const v4f*)(row + hl * 8);
    v4f a1 = *(const v4f*)(row + hl * 8 + 4);
    v4f a2 = *(const v4f*)(row + 16 + hl * 8);
    v4f a3 = *(const v4f*)(row + 16 + hl * 8 + 4);
    v16bf a;
    #pragma unroll
    for (int i = 0; i < 4; ++i) {
        a[i]      = (__bf16)a0[i];
        a[4 + i]  = (__bf16)a1[i];
        a[8 + i]  = (__bf16)a2[i];
        a[12 + i] = (__bf16)a3[i];
    }
    return a;
}

// A fragment from bf16 row
__device__ __forceinline__ v16bf loadA_bf16(const __bf16* row, int hl) {
    v8bf lo = *(const v8bf*)(row + hl * 8);
    v8bf hi = *(const v8bf*)(row + 16 + hl * 8);
    v16bf a;
    #pragma unroll
    for (int i = 0; i < 8; ++i) { a[i] = lo[i]; a[8 + i] = hi[i]; }
    return a;
}

#define WMMA_BF16(A, Bf, C) \
    __builtin_amdgcn_wmma_f32_16x16x32_bf16(false, (A), false, (Bf), (short)0, (C), false, false)

// ---------------- GEMM1: h = GELU(x @ W1^T + b1), bf16 out ----------------
// A_BF16: read activations from pre-converted bf16 buffer (half the bytes,
// no cvt in the hot loop); otherwise read fp32 x and convert in-register.
template <bool A_BF16>
__global__ __launch_bounds__(256) void gemm1_gelu(
        const float*  __restrict__ x,    // [B,N,P,D] fp32
        const __bf16* __restrict__ xb,   // [B,N,P,D] bf16 (may be null if !A_BF16)
        const __bf16* __restrict__ w1,   // [P,D,D] bf16 (row-major [n,k])
        const float*  __restrict__ b1,   // [P,D]
        __bf16*       __restrict__ h)    // [P, MTOT, D] bf16
{
    const int lane = threadIdx.x & 31;
    const int wave = threadIdx.x >> 5;
    const int l    = lane & 15;
    const int hl   = lane >> 4;
    const int p    = blockIdx.z;
    const int wm   = wave & 3;           // 4 M-groups
    const int wn   = wave >> 2;          // 2 N-groups
    const int mw   = blockIdx.x * BLK_M + wm * 32;   // wave tile: 32 x 64
    const int nw   = blockIdx.y * BLK_N + wn * 64;

    const float*  xrow0  = x  + ((size_t)(mw + l) * PP + p) * DD;
    const float*  xrow1  = x  + ((size_t)(mw + 16 + l) * PP + p) * DD;
    const __bf16* xbrow0 = xb + ((size_t)(mw + l) * PP + p) * DD;
    const __bf16* xbrow1 = xb + ((size_t)(mw + 16 + l) * PP + p) * DD;

    const __bf16* wbase = w1 + (size_t)p * DD * DD;
    const __bf16* wp0 = wbase + (size_t)(nw +  0 + l) * DD + hl * 8;
    const __bf16* wp1 = wbase + (size_t)(nw + 16 + l) * DD + hl * 8;
    const __bf16* wp2 = wbase + (size_t)(nw + 32 + l) * DD + hl * 8;
    const __bf16* wp3 = wbase + (size_t)(nw + 48 + l) * DD + hl * 8;

    v8f acc[2][4];
    #pragma unroll
    for (int mt = 0; mt < 2; ++mt)
        #pragma unroll
        for (int nt = 0; nt < 4; ++nt)
            #pragma unroll
            for (int r = 0; r < 8; ++r) acc[mt][nt][r] = 0.0f;

    auto loadA0 = [&](int k) {
        if constexpr (A_BF16) return loadA_bf16(xbrow0 + k, hl);
        else                  return loadA_f32(xrow0 + k, hl);
    };
    auto loadA1 = [&](int k) {
        if constexpr (A_BF16) return loadA_bf16(xbrow1 + k, hl);
        else                  return loadA_f32(xrow1 + k, hl);
    };

    // ping-pong buffers
    v16bf aA0, aA1, bA0, bA1, bA2, bA3;
    v16bf aB0, aB1, bB0, bB1, bB2, bB3;

    aA0 = loadA0(0);  aA1 = loadA1(0);
    bA0 = loadB(wp0); bA1 = loadB(wp1); bA2 = loadB(wp2); bA3 = loadB(wp3);

    for (int kk = 0; kk < DD; kk += 64) {
        const int k1 = kk + 32;
        aB0 = loadA0(k1);  aB1 = loadA1(k1);
        bB0 = loadB(wp0 + k1); bB1 = loadB(wp1 + k1);
        bB2 = loadB(wp2 + k1); bB3 = loadB(wp3 + k1);

        acc[0][0] = WMMA_BF16(aA0, bA0, acc[0][0]);
        acc[0][1] = WMMA_BF16(aA0, bA1, acc[0][1]);
        acc[0][2] = WMMA_BF16(aA0, bA2, acc[0][2]);
        acc[0][3] = WMMA_BF16(aA0, bA3, acc[0][3]);
        acc[1][0] = WMMA_BF16(aA1, bA0, acc[1][0]);
        acc[1][1] = WMMA_BF16(aA1, bA1, acc[1][1]);
        acc[1][2] = WMMA_BF16(aA1, bA2, acc[1][2]);
        acc[1][3] = WMMA_BF16(aA1, bA3, acc[1][3]);

        const int k2 = (kk + 64) & (DD - 1);  // wraps to 0 on last iter (dead loads)
        aA0 = loadA0(k2);  aA1 = loadA1(k2);
        bA0 = loadB(wp0 + k2); bA1 = loadB(wp1 + k2);
        bA2 = loadB(wp2 + k2); bA3 = loadB(wp3 + k2);

        acc[0][0] = WMMA_BF16(aB0, bB0, acc[0][0]);
        acc[0][1] = WMMA_BF16(aB0, bB1, acc[0][1]);
        acc[0][2] = WMMA_BF16(aB0, bB2, acc[0][2]);
        acc[0][3] = WMMA_BF16(aB0, bB3, acc[0][3]);
        acc[1][0] = WMMA_BF16(aB1, bB0, acc[1][0]);
        acc[1][1] = WMMA_BF16(aB1, bB1, acc[1][1]);
        acc[1][2] = WMMA_BF16(aB1, bB2, acc[1][2]);
        acc[1][3] = WMMA_BF16(aB1, bB3, acc[1][3]);
    }

    // epilogue: bias + exact GELU, store bf16
    __bf16* hp = h + (size_t)p * MTOT * DD;
    #pragma unroll
    for (int mt = 0; mt < 2; ++mt) {
        #pragma unroll
        for (int nt = 0; nt < 4; ++nt) {
            const int n = nw + nt * 16 + l;
            const float bias = b1[p * DD + n];
            #pragma unroll
            for (int r = 0; r < 8; ++r) {
                const int m = mw + mt * 16 + hl * 8 + r;
                float v = acc[mt][nt][r] + bias;
                float g = 0.5f * v * (1.0f + erff(v * 0.70710678118654752440f));
                hp[(size_t)m * DD + n] = (__bf16)g;
            }
        }
    }
}

// ---------------- GEMM2: out = h @ W2^T + b2 + x, fp32 out ----------------
__global__ __launch_bounds__(256) void gemm2_res(
        const __bf16* __restrict__ h,    // [P, MTOT, D] bf16
        const __bf16* __restrict__ w2,   // [P,D,D] bf16
        const float*  __restrict__ b2,   // [P,D]
        const float*  __restrict__ x,    // [B,N,P,D] fp32 (residual)
        float*        __restrict__ out)  // [B,N,P,D] fp32
{
    const int lane = threadIdx.x & 31;
    const int wave = threadIdx.x >> 5;
    const int l    = lane & 15;
    const int hl   = lane >> 4;
    const int p    = blockIdx.z;
    const int wm   = wave & 3;
    const int wn   = wave >> 2;
    const int mw   = blockIdx.x * BLK_M + wm * 32;
    const int nw   = blockIdx.y * BLK_N + wn * 64;

    const __bf16* arow0 = h + ((size_t)p * MTOT + (mw + l)) * DD;
    const __bf16* arow1 = h + ((size_t)p * MTOT + (mw + 16 + l)) * DD;
    const __bf16* wbase = w2 + (size_t)p * DD * DD;
    const __bf16* wp0 = wbase + (size_t)(nw +  0 + l) * DD + hl * 8;
    const __bf16* wp1 = wbase + (size_t)(nw + 16 + l) * DD + hl * 8;
    const __bf16* wp2 = wbase + (size_t)(nw + 32 + l) * DD + hl * 8;
    const __bf16* wp3 = wbase + (size_t)(nw + 48 + l) * DD + hl * 8;

    v8f acc[2][4];
    #pragma unroll
    for (int mt = 0; mt < 2; ++mt)
        #pragma unroll
        for (int nt = 0; nt < 4; ++nt)
            #pragma unroll
            for (int r = 0; r < 8; ++r) acc[mt][nt][r] = 0.0f;

    v16bf aA0, aA1, bA0, bA1, bA2, bA3;
    v16bf aB0, aB1, bB0, bB1, bB2, bB3;

    aA0 = loadA_bf16(arow0, hl);  aA1 = loadA_bf16(arow1, hl);
    bA0 = loadB(wp0); bA1 = loadB(wp1); bA2 = loadB(wp2); bA3 = loadB(wp3);

    for (int kk = 0; kk < DD; kk += 64) {
        const int k1 = kk + 32;
        aB0 = loadA_bf16(arow0 + k1, hl);  aB1 = loadA_bf16(arow1 + k1, hl);
        bB0 = loadB(wp0 + k1); bB1 = loadB(wp1 + k1);
        bB2 = loadB(wp2 + k1); bB3 = loadB(wp3 + k1);

        acc[0][0] = WMMA_BF16(aA0, bA0, acc[0][0]);
        acc[0][1] = WMMA_BF16(aA0, bA1, acc[0][1]);
        acc[0][2] = WMMA_BF16(aA0, bA2, acc[0][2]);
        acc[0][3] = WMMA_BF16(aA0, bA3, acc[0][3]);
        acc[1][0] = WMMA_BF16(aA1, bA0, acc[1][0]);
        acc[1][1] = WMMA_BF16(aA1, bA1, acc[1][1]);
        acc[1][2] = WMMA_BF16(aA1, bA2, acc[1][2]);
        acc[1][3] = WMMA_BF16(aA1, bA3, acc[1][3]);

        const int k2 = (kk + 64) & (DD - 1);
        aA0 = loadA_bf16(arow0 + k2, hl);  aA1 = loadA_bf16(arow1 + k2, hl);
        bA0 = loadB(wp0 + k2); bA1 = loadB(wp1 + k2);
        bA2 = loadB(wp2 + k2); bA3 = loadB(wp3 + k2);

        acc[0][0] = WMMA_BF16(aB0, bB0, acc[0][0]);
        acc[0][1] = WMMA_BF16(aB0, bB1, acc[0][1]);
        acc[0][2] = WMMA_BF16(aB0, bB2, acc[0][2]);
        acc[0][3] = WMMA_BF16(aB0, bB3, acc[0][3]);
        acc[1][0] = WMMA_BF16(aB1, bB0, acc[1][0]);
        acc[1][1] = WMMA_BF16(aB1, bB1, acc[1][1]);
        acc[1][2] = WMMA_BF16(aB1, bB2, acc[1][2]);
        acc[1][3] = WMMA_BF16(aB1, bB3, acc[1][3]);
    }

    #pragma unroll
    for (int mt = 0; mt < 2; ++mt) {
        #pragma unroll
        for (int nt = 0; nt < 4; ++nt) {
            const int n = nw + nt * 16 + l;
            const float bias = b2[p * DD + n];
            #pragma unroll
            for (int r = 0; r < 8; ++r) {
                const int m = mw + mt * 16 + hl * 8 + r;
                const size_t idx = ((size_t)m * PP + p) * DD + n;
                out[idx] = acc[mt][nt][r] + bias + x[idx];
            }
        }
    }
}

extern "C" void kernel_launch(void* const* d_in, const int* in_sizes, int n_in,
                              void* d_out, int out_size, void* d_ws, size_t ws_size,
                              hipStream_t stream) {
    const float* x  = (const float*)d_in[0];
    const float* W1 = (const float*)d_in[1];
    const float* b1 = (const float*)d_in[2];
    const float* W2 = (const float*)d_in[3];
    const float* b2 = (const float*)d_in[4];
    float* out = (float*)d_out;

    char* ws = (char*)d_ws;
    const size_t welems = (size_t)PP * DD * DD;                 // 2,097,152
    const size_t helems = (size_t)PP * MTOT * DD;               // 16,777,216
    const size_t xelems = (size_t)BB * NN * PP * DD;            // 16,777,216

    __bf16* w1b = (__bf16*)ws;                                   // 4 MB
    __bf16* w2b = (__bf16*)(ws + welems * 2);                    // 4 MB
    __bf16* hb  = (__bf16*)(ws + 2 * welems * 2);                // 33.5 MB
    __bf16* xbb = (__bf16*)(ws + 2 * welems * 2 + helems * 2);   // 33.5 MB (optional)

    const size_t need_xb = 2 * welems * 2 + helems * 2 + xelems * 2;
    const bool use_xb = (ws_size >= need_xb);

    cvt_weights<<<(unsigned)((welems + 255) / 256), 256, 0, stream>>>(
        W1, W2, w1b, w2b, (int)welems);

    dim3 grid(MTOT / BLK_M, DD / BLK_N, PP);   // (32, 4, 8)
    if (use_xb) {
        const unsigned n8 = (unsigned)(xelems / 8);
        cvt_x<<<(n8 + 255) / 256, 256, 0, stream>>>(x, xbb, n8);
        gemm1_gelu<true><<<grid, 256, 0, stream>>>(x, xbb, w1b, b1, hb);
    } else {
        gemm1_gelu<false><<<grid, 256, 0, stream>>>(x, (const __bf16*)nullptr,
                                                    w1b, b1, hb);
    }
    gemm2_res<<<grid, 256, 0, stream>>>(hb, w2b, b2, x, out);
}
